// GraphTransformer_54039278518841
// MI455X (gfx1250) — compile-verified, matrix operand order checked by reading, not run on hardware
//
#include <hip/hip_runtime.h>
#include <hip/hip_bf16.h>
#include <math.h>
#include <stdint.h>

// Problem constants (from reference)
#define N_NODES 50000
#define N_EDGES 800000
#define IN_C    128
#define D1C     256   // heads*hid after layer-1 concat
#define H1      4
#define HD1     64
#define EMB     128

typedef __attribute__((ext_vector_type(16))) __bf16  bf16x16;
typedef __attribute__((ext_vector_type(8)))  float   f32x8;
typedef unsigned int u32x4 __attribute__((ext_vector_type(4)));
typedef int          i32x8 __attribute__((ext_vector_type(8)));
typedef int          i32x4 __attribute__((ext_vector_type(4)));

__device__ __forceinline__ unsigned short f2bf(float f) {
  unsigned u = __float_as_uint(f);
  unsigned r = 0x7FFFu + ((u >> 16) & 1u);   // round-to-nearest-even
  return (unsigned short)((u + r) >> 16);
}
// order-preserving float <-> uint key (for atomicMax-based segment max)
__device__ __forceinline__ unsigned fkey(float f) {
  unsigned u = __float_as_uint(f);
  return (u & 0x80000000u) ? ~u : (u | 0x80000000u);
}
__device__ __forceinline__ float funkey(unsigned k) {
  unsigned u = (k & 0x80000000u) ? (k ^ 0x80000000u) : ~k;
  return __uint_as_float(u);
}

// ---- Tensor Data Mover: 1-D bulk copy global -> LDS (64 KB weight stage) ---
// D# per CDNA5 ISA ch.8: group0 = {flags, lds_addr, global_addr, type=2},
// group1 = dims/strides for a 1-D tensor of `ndw` dwords, tile = whole tensor.
__device__ __forceinline__ void tdm_load_to_lds(const void* gsrc,
                                                unsigned lds_byte_off,
                                                unsigned nbytes) {
#if __has_builtin(__builtin_amdgcn_tensor_load_to_lds)
  unsigned ndw = nbytes >> 2;                       // dword elements
  unsigned long long ga = (unsigned long long)(uintptr_t)gsrc;
  u32x4 g0;
  g0[0] = 1u;                                        // count=1, no gather
  g0[1] = lds_byte_off;                              // lds_addr
  g0[2] = (unsigned)(ga & 0xFFFFFFFFull);            // global_addr[31:0]
  g0[3] = (unsigned)((ga >> 32) & 0x1FFFFFFull)      // global_addr[56:32]
        | (2u << 30);                                // type = 2 ("image")
  i32x8 g1;
  g1[0] = (int)(2u << 16);                           // data_size = 4 bytes
  g1[1] = (int)((ndw & 0xFFFFu) << 16);              // tensor_dim0[15:0]
  g1[2] = (int)(((ndw >> 16) & 0xFFFFu) | (1u << 16)); // dim0 hi | tensor_dim1=1
  g1[3] = (int)((ndw & 0xFFFFu) << 16);              // tile_dim0 = ndw
  g1[4] = 1;                                         // tile_dim1 = 1
  g1[5] = (int)ndw;                                  // tensor_dim0_stride lo32
  g1[6] = (int)((ndw & 0xFFFFu) << 16);              // dim1_stride[15:0]
  g1[7] = 0;
  i32x4 z4 = {0, 0, 0, 0};
#if defined(__clang_major__) && (__clang_major__ >= 23)
  i32x8 z8 = {0, 0, 0, 0, 0, 0, 0, 0};
  __builtin_amdgcn_tensor_load_to_lds(g0, g1, z4, z4, z8, 0);
#else
  __builtin_amdgcn_tensor_load_to_lds(g0, g1, z4, z4, 0);
#endif
  __builtin_amdgcn_s_wait_tensorcnt(0);
#else
  (void)gsrc; (void)lds_byte_off; (void)nbytes;
#endif
}

// ---------------- fp32 -> bf16 (same row-major layout) ----------------------
__global__ void k_f32_to_bf16(const float* __restrict__ in,
                              unsigned short* __restrict__ out, int n) {
  int i = blockIdx.x * blockDim.x + threadIdx.x;
  if (i < n) out[i] = f2bf(in[i]);
}

// ------- pack W[K,Nw] (row-major fp32) into WMMA B-fragment bf16 order ------
// tile (nt,kt), lane L holds 16 bf16: n = nt*16+(L&15), k = kt*32+(L>>4)*16+t
// pack offset = ((nt*ktiles + kt)*32 + L)*16 + t   (32B contiguous per lane)
__global__ void k_pack_w(const float* __restrict__ W,
                         unsigned short* __restrict__ Wp, int K, int Nw) {
  int tid = blockIdx.x * blockDim.x + threadIdx.x;
  int ktiles = K >> 5;
  int total = (Nw >> 4) * ktiles * 32;
  if (tid >= total) return;
  int lane = tid & 31;
  int n  = ((tid / (32 * ktiles)) << 4) + (lane & 15);
  int kb = (((tid >> 5) % ktiles) << 5) + ((lane >> 4) << 4);
  unsigned short* dst = Wp + (size_t)tid * 16;
#pragma unroll
  for (int t = 0; t < 16; ++t)
    dst[t] = f2bf(W[(size_t)(kb + t) * Nw + n]);
}

// --------------- bf16 WMMA GEMM: C[M,Nw] = A[M,K] * W + bias ----------------
// Packed weight (64 KB) is staged into LDS once per block via the TDM, then
// every wave loops over 16x64 C tiles: A fragments from global (contiguous
// 32B/lane), B fragments from LDS, 4 WMMAs per k-step.
__global__ void __launch_bounds__(256)
k_gemm_bf16(const unsigned short* __restrict__ Abf,
            const unsigned short* __restrict__ Wp,
            const float* __restrict__ bias,
            float* __restrict__ C, int M, int K, int Nw) {
  __shared__ unsigned short ldsW[32768];            // 64 KB = K*Nw bf16
  const int ktiles = K >> 5;
  const int wbytes = K * Nw * 2;                    // 65536 for both layers
  const int wavein = threadIdx.x >> 5;
  const int lane   = threadIdx.x & 31;

#if __has_builtin(__builtin_amdgcn_tensor_load_to_lds)
  if (wavein == 0) {                                // one TDM op per block
    tdm_load_to_lds(Wp, 0u /*static LDS starts at 0*/, (unsigned)wbytes);
  }
#else
  for (int i = threadIdx.x; i < wbytes / 16; i += blockDim.x)
    ((uint4*)ldsW)[i] = ((const uint4*)Wp)[i];
#endif
  // The TDM writes LDS behind the compiler's back; escape the LDS object and
  // clobber memory so reads of ldsW cannot be folded to undef / DCE'd.
  asm volatile("" : : "r"(&ldsW[0]) : "memory");
  __syncthreads();

  const int ngroups = Nw >> 6;
  const int tiles   = (M >> 4) * ngroups;
  const int stride  = (int)gridDim.x * 8;
  const size_t bstride = (size_t)ktiles * 32 * 16;  // one 16-wide n-tile

  for (int tile = (int)blockIdx.x * 8 + wavein; tile < tiles; tile += stride) {
    int mt = tile / ngroups;
    int ng = tile - mt * ngroups;
    f32x8 acc0 = {}, acc1 = {}, acc2 = {}, acc3 = {};
    // A fragment: lane {m = L&15, k-half = L>>4} -> 16 contiguous bf16
    const unsigned short* Arow =
        Abf + (size_t)(mt * 16 + (lane & 15)) * K + ((lane >> 4) << 4);
    const unsigned short* Bbase =
        ldsW + ((size_t)(ng * 4) * ktiles * 32 + (size_t)lane) * 16;
    for (int kt = 0; kt < ktiles; ++kt) {
      bf16x16 a = *(const bf16x16*)(Arow + kt * 32);
      const unsigned short* bp = Bbase + (size_t)kt * 32 * 16;
      bf16x16 b0 = *(const bf16x16*)(bp);
      bf16x16 b1 = *(const bf16x16*)(bp + bstride);
      bf16x16 b2 = *(const bf16x16*)(bp + 2 * bstride);
      bf16x16 b3 = *(const bf16x16*)(bp + 3 * bstride);
      acc0 = __builtin_amdgcn_wmma_f32_16x16x32_bf16(false, a, false, b0, (short)0, acc0, false, false);
      acc1 = __builtin_amdgcn_wmma_f32_16x16x32_bf16(false, a, false, b1, (short)0, acc1, false, false);
      acc2 = __builtin_amdgcn_wmma_f32_16x16x32_bf16(false, a, false, b2, (short)0, acc2, false, false);
      acc3 = __builtin_amdgcn_wmma_f32_16x16x32_bf16(false, a, false, b3, (short)0, acc3, false, false);
    }
    // C layout: VGPR r, lanes 0-15 -> row 8*(L>>4)+r, col = lane
    int col  = ng * 64 + (lane & 15);
    int row0 = mt * 16 + ((lane >> 4) << 3);
#pragma unroll
    for (int r = 0; r < 8; ++r) {
      float* crow = C + (size_t)(row0 + r) * Nw + col;
      crow[0]  = acc0[r] + bias[col];
      crow[16] = acc1[r] + bias[col + 16];
      crow[32] = acc2[r] + bias[col + 32];
      crow[48] = acc3[r] + bias[col + 48];
    }
  }
}

// ---------------- edge logits + segment max (atomic, keyed) -----------------
__global__ void k_edge_logits(const float* __restrict__ Q, const float* __restrict__ Kf,
                              const long long* __restrict__ src, const long long* __restrict__ dst,
                              float* __restrict__ logits, unsigned* __restrict__ maxbits,
                              int E, int H, int D, float scale) {
  int idx = blockIdx.x * blockDim.x + threadIdx.x;
  if (idx >= E * H) return;
  int e = idx / H, h = idx - e * H;
  size_t s = (size_t)src[e], d = (size_t)dst[e];
  const float4* qp = (const float4*)(Q  + (d * H + h) * D);
  const float4* kp = (const float4*)(Kf + (s * H + h) * D);
  float acc = 0.f;
#pragma unroll 4
  for (int i = 0; i < D / 4; ++i) {
    float4 a = qp[i], b = kp[i];
    acc += a.x * b.x + a.y * b.y + a.z * b.z + a.w * b.w;
  }
  acc *= scale;
  logits[idx] = acc;
  atomicMax(&maxbits[d * H + h], fkey(acc));
}

// ---------------- exp(l - max) + segment sum --------------------------------
__global__ void k_edge_exp(float* __restrict__ logits,
                           const unsigned* __restrict__ maxbits,
                           float* __restrict__ sum,
                           const long long* __restrict__ dst, int E, int H) {
  int idx = blockIdx.x * blockDim.x + threadIdx.x;
  if (idx >= E * H) return;
  int e = idx / H, h = idx - e * H;
  size_t d = (size_t)dst[e];
  float ex = __expf(logits[idx] - funkey(maxbits[d * H + h]));
  logits[idx] = ex;
  atomicAdd(&sum[d * H + h], ex);
}

// ---------------- scatter alpha * v[src] into out[dst] ----------------------
__global__ void k_edge_scatter(const float* __restrict__ V,
                               const float* __restrict__ ex,
                               const float* __restrict__ sum,
                               const long long* __restrict__ src,
                               const long long* __restrict__ dst,
                               float* __restrict__ out, int E, int H, int D) {
  int chunks = D >> 2;
  long long idx = (long long)blockIdx.x * blockDim.x + threadIdx.x;
  if (idx >= (long long)E * H * chunks) return;
  int c  = (int)(idx % chunks);
  int eh = (int)(idx / chunks);
  int e = eh / H, h = eh - e * H;
  size_t s = (size_t)src[e], d = (size_t)dst[e];
  float alpha = ex[eh] / sum[d * H + h];
  float4 v = ((const float4*)(V + (s * H + h) * D))[c];
  float* o = out + (d * H + h) * D + c * 4;
  atomicAdd(o + 0, alpha * v.x);
  atomicAdd(o + 1, alpha * v.y);
  atomicAdd(o + 2, alpha * v.z);
  atomicAdd(o + 3, alpha * v.w);
}

// ---------------- skip + ReLU, emit bf16 activations ------------------------
__global__ void k_combine_relu_bf16(const float* __restrict__ attn,
                                    const float* __restrict__ skip,
                                    unsigned short* __restrict__ hbf, int n) {
  int i = blockIdx.x * blockDim.x + threadIdx.x;
  if (i >= n) return;
  float v = attn[i] + skip[i];
  hbf[i] = f2bf(v > 0.f ? v : 0.f);
}

// ---------------- final mean over nodes -------------------------------------
__global__ void k_final_reduce(const float* __restrict__ attn,
                               const float* __restrict__ skip,
                               float* __restrict__ acc, int n) {
  int c = threadIdx.x;                 // 128 channels
  float s = 0.f;
  for (int node = blockIdx.x; node < n; node += gridDim.x)
    s += attn[(size_t)node * EMB + c] + skip[(size_t)node * EMB + c];
  atomicAdd(&acc[c], s);
}
__global__ void k_final_scale(const float* __restrict__ acc,
                              float* __restrict__ out, float invN) {
  out[threadIdx.x] = acc[threadIdx.x] * invN;
}

static inline int blocks_for(long long n, int bs) { return (int)((n + bs - 1) / bs); }

extern "C" void kernel_launch(void* const* d_in, const int* in_sizes, int n_in,
                              void* d_out, int out_size, void* d_ws, size_t ws_size,
                              hipStream_t stream) {
  (void)in_sizes; (void)n_in; (void)out_size; (void)ws_size;
  const float*     x   = (const float*)d_in[0];
  const long long* ei  = (const long long*)d_in[1];
  const float *Wq1 = (const float*)d_in[2],  *bq1 = (const float*)d_in[3];
  const float *Wk1 = (const float*)d_in[4],  *bk1 = (const float*)d_in[5];
  const float *Wv1 = (const float*)d_in[6],  *bv1 = (const float*)d_in[7];
  const float *Ws1 = (const float*)d_in[8],  *bs1 = (const float*)d_in[9];
  const float *Wq2 = (const float*)d_in[10], *bq2 = (const float*)d_in[11];
  const float *Wk2 = (const float*)d_in[12], *bk2 = (const float*)d_in[13];
  const float *Wv2 = (const float*)d_in[14], *bv2 = (const float*)d_in[15];
  const float *Ws2 = (const float*)d_in[16], *bs2 = (const float*)d_in[17];
  float* out = (float*)d_out;
  float* ws  = (float*)d_ws;

  const long long* src = ei;            // edge_index[0]
  const long long* dst = ei + N_EDGES;  // edge_index[1]

  // workspace layout (float units)
  const size_t offA    = 0;                                   // N*1024 (Q/K/V/S both layers, overlaid)
  const size_t offOut1 = offA    + (size_t)N_NODES * 1024;    // N*256 attn accum L1
  const size_t offLog  = offOut1 + (size_t)N_NODES * 256;     // E*4 logits / exp
  const size_t offMax  = offLog  + (size_t)N_EDGES * 4;       // N*4 keyed max (uint)
  const size_t offSum  = offMax  + (size_t)N_NODES * 4;       // N*4 softmax denom
  const size_t offXbf  = offSum  + (size_t)N_NODES * 4;       // N*128 bf16 = N*64 f
  const size_t offHbf  = offXbf  + (size_t)N_NODES * 64;      // N*256 bf16 = N*128 f
  const size_t offWp   = offHbf  + (size_t)N_NODES * 128;     // 8 * 16384 f packed W
  const size_t offAcc  = offWp   + 8 * 16384;                 // 128 f

  float* Q1 = ws + offA;
  float* K1 = Q1 + (size_t)N_NODES * D1C;
  float* V1 = K1 + (size_t)N_NODES * D1C;
  float* S1 = V1 + (size_t)N_NODES * D1C;
  float* out1 = ws + offOut1;
  float* logit = ws + offLog;
  unsigned* maxb = (unsigned*)(ws + offMax);
  float* sum = ws + offSum;
  unsigned short* xbf = (unsigned short*)(ws + offXbf);
  unsigned short* hbf = (unsigned short*)(ws + offHbf);
  unsigned short* wp  = (unsigned short*)(ws + offWp);
  float* acc = ws + offAcc;
  // layer-2 overlay of region A (layer-1 tensors dead by then)
  float* Q2 = ws + offA;
  float* K2 = Q2 + (size_t)N_NODES * EMB;
  float* V2 = K2 + (size_t)N_NODES * EMB;
  float* S2 = V2 + (size_t)N_NODES * EMB;
  float* out2 = S2 + (size_t)N_NODES * EMB;

  const size_t wp1 = (size_t)IN_C * D1C;   // 32768 bf16 per layer-1 matrix
  const size_t wp2 = (size_t)D1C * EMB;    // 32768 bf16 per layer-2 matrix
  unsigned short* Wpq1 = wp;
  unsigned short* Wpk1 = Wpq1 + wp1;
  unsigned short* Wpv1 = Wpk1 + wp1;
  unsigned short* Wps1 = Wpv1 + wp1;
  unsigned short* Wpq2 = Wps1 + wp1;
  unsigned short* Wpk2 = Wpq2 + wp2;
  unsigned short* Wpv2 = Wpk2 + wp2;
  unsigned short* Wps2 = Wpv2 + wp2;

  // ---- init ----
  hipMemsetAsync(out1, 0, (size_t)N_NODES * 256 * 4, stream);
  hipMemsetAsync(maxb, 0, (size_t)N_NODES * 4 * 4, stream);
  hipMemsetAsync(sum,  0, (size_t)N_NODES * 4 * 4, stream);
  hipMemsetAsync(acc,  0, 128 * 4, stream);

  // ---- weight packing (bf16 fragment order) ----
  {
    int t1 = (D1C / 16) * (IN_C / 32) * 32;  // 2048
    int t2 = (EMB / 16) * (D1C / 32) * 32;   // 2048
    k_pack_w<<<blocks_for(t1, 256), 256, 0, stream>>>(Wq1, Wpq1, IN_C, D1C);
    k_pack_w<<<blocks_for(t1, 256), 256, 0, stream>>>(Wk1, Wpk1, IN_C, D1C);
    k_pack_w<<<blocks_for(t1, 256), 256, 0, stream>>>(Wv1, Wpv1, IN_C, D1C);
    k_pack_w<<<blocks_for(t1, 256), 256, 0, stream>>>(Ws1, Wps1, IN_C, D1C);
    k_pack_w<<<blocks_for(t2, 256), 256, 0, stream>>>(Wq2, Wpq2, D1C, EMB);
    k_pack_w<<<blocks_for(t2, 256), 256, 0, stream>>>(Wk2, Wpk2, D1C, EMB);
    k_pack_w<<<blocks_for(t2, 256), 256, 0, stream>>>(Wv2, Wpv2, D1C, EMB);
    k_pack_w<<<blocks_for(t2, 256), 256, 0, stream>>>(Ws2, Wps2, D1C, EMB);
  }

  // ---- layer 1: activations to bf16, 4 WMMA GEMMs ----
  k_f32_to_bf16<<<blocks_for((long long)N_NODES * IN_C, 256), 256, 0, stream>>>(
      x, xbf, N_NODES * IN_C);
  {
    const int blks = 192;   // persistent-ish: each block stages W once via TDM
    k_gemm_bf16<<<blks, 256, 0, stream>>>(xbf, Wpq1, bq1, Q1, N_NODES, IN_C, D1C);
    k_gemm_bf16<<<blks, 256, 0, stream>>>(xbf, Wpk1, bk1, K1, N_NODES, IN_C, D1C);
    k_gemm_bf16<<<blks, 256, 0, stream>>>(xbf, Wpv1, bv1, V1, N_NODES, IN_C, D1C);
    k_gemm_bf16<<<blks, 256, 0, stream>>>(xbf, Wps1, bs1, S1, N_NODES, IN_C, D1C);
  }

  // ---- layer 1 edge attention ----
  {
    const float scale1 = 0.125f;  // 1/sqrt(64)
    int eh = N_EDGES * H1;
    k_edge_logits<<<blocks_for(eh, 256), 256, 0, stream>>>(
        Q1, K1, src, dst, logit, maxb, N_EDGES, H1, HD1, scale1);
    k_edge_exp<<<blocks_for(eh, 256), 256, 0, stream>>>(
        logit, maxb, sum, dst, N_EDGES, H1);
    long long sc = (long long)eh * (HD1 / 4);
    k_edge_scatter<<<blocks_for(sc, 256), 256, 0, stream>>>(
        V1, logit, sum, src, dst, out1, N_EDGES, H1, HD1);
  }
  k_combine_relu_bf16<<<blocks_for((long long)N_NODES * D1C, 256), 256, 0, stream>>>(
      out1, S1, hbf, N_NODES * D1C);

  // ---- layer 2 ----
  hipMemsetAsync(maxb, 0, (size_t)N_NODES * 4, stream);
  hipMemsetAsync(sum,  0, (size_t)N_NODES * 4, stream);
  hipMemsetAsync(out2, 0, (size_t)N_NODES * EMB * 4, stream);
  {
    const int blks = 192;
    k_gemm_bf16<<<blks, 256, 0, stream>>>(hbf, Wpq2, bq2, Q2, N_NODES, D1C, EMB);
    k_gemm_bf16<<<blks, 256, 0, stream>>>(hbf, Wpk2, bk2, K2, N_NODES, D1C, EMB);
    k_gemm_bf16<<<blks, 256, 0, stream>>>(hbf, Wpv2, bv2, V2, N_NODES, D1C, EMB);
    k_gemm_bf16<<<blks, 256, 0, stream>>>(hbf, Wps2, bs2, S2, N_NODES, D1C, EMB);
  }
  {
    const float scale2 = 0.08838834764831845f;  // 1/sqrt(128)
    k_edge_logits<<<blocks_for(N_EDGES, 256), 256, 0, stream>>>(
        Q2, K2, src, dst, logit, maxb, N_EDGES, 1, EMB, scale2);
    k_edge_exp<<<blocks_for(N_EDGES, 256), 256, 0, stream>>>(
        logit, maxb, sum, dst, N_EDGES, 1);
    long long sc = (long long)N_EDGES * (EMB / 4);
    k_edge_scatter<<<blocks_for(sc, 256), 256, 0, stream>>>(
        V2, logit, sum, src, dst, out2, N_EDGES, 1, EMB);
  }

  // ---- mean over nodes -> [1,128] ----
  k_final_reduce<<<256, 128, 0, stream>>>(out2, S2, acc, N_NODES);
  k_final_scale<<<1, 128, 0, stream>>>(acc, out, 1.0f / (float)N_NODES);
}
